// MultiHeadAttention_41223096107098
// MI455X (gfx1250) — matrix operand order repaired and checked by reading, not compiled, not run on hardware
//
#include <hip/hip_runtime.h>

#define Bb 4
#define Ss 2048
#define Ee 1024
#define Hh 16
#define Dd 64

typedef __attribute__((ext_vector_type(16))) __bf16 v16bf;
typedef __attribute__((ext_vector_type(8)))  float  v8f;

union FragB { v16bf v; uint4 q[2]; __bf16 e[16]; };
union FragF { v8f v; float f[8]; };

__device__ __forceinline__ v8f wmma_bf16(v16bf a, v16bf b, v8f c) {
  return __builtin_amdgcn_wmma_f32_16x16x32_bf16(false, a, false, b, (short)0, c,
                                                 false, false);
}

// ---------------------------------------------------------------------------
// Weight convert+transpose: W fp32 [K=1024, N=1024] -> Wt bf16 [N][K]
// ---------------------------------------------------------------------------
__global__ __launch_bounds__(256) void wt_kernel(const float* __restrict__ W,
                                                 __bf16* __restrict__ Wt) {
  const int idx = blockIdx.x * 256 + threadIdx.x;   // 1M elements
  const int n = idx & (Ee - 1);
  const int k = idx >> 10;
  Wt[(size_t)n * Ee + k] = (__bf16)W[(size_t)k * Ee + n];
}

// ---------------------------------------------------------------------------
// Projection GEMM: Cout(bf16)[8192,1024] = A(f32) @ Wt^T + bias
// 4 waves / block; each wave: 16 rows x 64 cols, K-loop in steps of 32.
// ---------------------------------------------------------------------------
__global__ __launch_bounds__(128) void gemm_qkv_kernel(
    const float* __restrict__ A, const __bf16* __restrict__ Wt,
    const float* __restrict__ bias, __bf16* __restrict__ Cout)
{
  const int lane = threadIdx.x & 31;
  const int wave = threadIdx.x >> 5;
  const int l15  = lane & 15;
  const int hi   = lane >> 4;
  const int m0 = blockIdx.x * 64 + wave * 16;
  const int n0 = blockIdx.y * 64;

  const float* arow = A + (size_t)(m0 + l15) * Ee;

  FragF acc[4];
  #pragma unroll
  for (int c = 0; c < 4; ++c)
    #pragma unroll
    for (int i = 0; i < 8; ++i) acc[c].f[i] = 0.f;

  for (int kk = 0; kk < Ee; kk += 32) {
    // A fragment: lane<16 holds K {0..7,16..23}+kk, lane>=16 {8..15,24..31}+kk
    FragB af;
    {
      const float* p = arow + kk + hi * 8;
      float tmp[16] __attribute__((aligned(16)));
      *(float4*)&tmp[0]  = *(const float4*)(p + 0);
      *(float4*)&tmp[4]  = *(const float4*)(p + 4);
      *(float4*)&tmp[8]  = *(const float4*)(p + 16);
      *(float4*)&tmp[12] = *(const float4*)(p + 20);
      #pragma unroll
      for (int i = 0; i < 16; ++i) af.e[i] = (__bf16)tmp[i];   // native cvt
    }
    #pragma unroll
    for (int c = 0; c < 4; ++c) {
      // B fragment: lane holds col n0+c*16+l15, K {0..7,16..23}+kk (lane<16) / {8..15,24..31}+kk
      FragB wb;
      const __bf16* wp = Wt + (size_t)(n0 + c * 16 + l15) * Ee + kk + hi * 8;
      wb.q[0] = *(const uint4*)(wp);
      wb.q[1] = *(const uint4*)(wp + 16);
      acc[c].v = wmma_bf16(af.v, wb.v, acc[c].v);
    }
  }

  #pragma unroll
  for (int c = 0; c < 4; ++c) {
    const int col = n0 + c * 16 + l15;
    const float bv = bias[col];
    #pragma unroll
    for (int r = 0; r < 8; ++r) {
      const int mrow = m0 + r + hi * 8;
      Cout[(size_t)mrow * Ee + col] = (__bf16)(acc[c].f[r] + bv);
    }
  }
}

// ---------------------------------------------------------------------------
// Output GEMM: Cout(f32) = A(bf16) @ Wt^T + bias
// ---------------------------------------------------------------------------
__global__ __launch_bounds__(128) void gemm_out_kernel(
    const __bf16* __restrict__ A, const __bf16* __restrict__ Wt,
    const float* __restrict__ bias, float* __restrict__ Cout)
{
  const int lane = threadIdx.x & 31;
  const int wave = threadIdx.x >> 5;
  const int l15  = lane & 15;
  const int hi   = lane >> 4;
  const int m0 = blockIdx.x * 64 + wave * 16;
  const int n0 = blockIdx.y * 64;

  const __bf16* arow = A + (size_t)(m0 + l15) * Ee;

  FragF acc[4];
  #pragma unroll
  for (int c = 0; c < 4; ++c)
    #pragma unroll
    for (int i = 0; i < 8; ++i) acc[c].f[i] = 0.f;

  for (int kk = 0; kk < Ee; kk += 32) {
    FragB af;
    {
      const __bf16* p = arow + kk + hi * 8;
      af.q[0] = *(const uint4*)(p);
      af.q[1] = *(const uint4*)(p + 16);
    }
    #pragma unroll
    for (int c = 0; c < 4; ++c) {
      FragB wb;
      const __bf16* wp = Wt + (size_t)(n0 + c * 16 + l15) * Ee + kk + hi * 8;
      wb.q[0] = *(const uint4*)(wp);
      wb.q[1] = *(const uint4*)(wp + 16);
      acc[c].v = wmma_bf16(af.v, wb.v, acc[c].v);
    }
  }

  #pragma unroll
  for (int c = 0; c < 4; ++c) {
    const int col = n0 + c * 16 + l15;
    const float bv = bias[col];
    #pragma unroll
    for (int r = 0; r < 8; ++r) {
      const int mrow = m0 + r + hi * 8;
      Cout[(size_t)mrow * Ee + col] = acc[c].f[r] + bv;
    }
  }
}

// ---------------------------------------------------------------------------
// Fused flash-style attention.
// Grid (S/128, H, B), block 128 (4 waves); wave owns TWO 16-query tiles
// (32 rows) so each K/V fragment load feeds 2 WMMAs.
// Pass 1: online row max + sum-exp (no stores).
// Pass 2: recompute logits, write normalized attn once, WMMA-accumulate ctx.
// Logits computed transposed: C[key,q] = K_tile(A) x Q^T(B) so that
//  - A/B fragments are contiguous 16B loads from global memory,
//  - each lane holds a full q-row slice (softmax = 1 shfl_xor(16)),
//  - probability C-tiles re-pack per-lane straight into the ctx WMMA A-fragment.
// V tile is transposed through LDS so B_v fragments are contiguous ds_load_b128.
// ---------------------------------------------------------------------------
__global__ __launch_bounds__(128) void attn_kernel(
    const __bf16* __restrict__ Qh, const __bf16* __restrict__ Kh,
    const __bf16* __restrict__ Vh, float* __restrict__ attn,
    __bf16* __restrict__ ctxout)
{
  __shared__ __bf16 Vt[Dd][32];   // [d][key] for current 32-key tile

  const int lane = threadIdx.x & 31;
  const int wave = threadIdx.x >> 5;
  const int l15  = lane & 15;
  const int hi   = lane >> 4;
  const int b = blockIdx.z;
  const int h = blockIdx.y;
  const int q0 = blockIdx.x * 128 + wave * 32;   // 32 q-rows per wave

  // Q as B-fragments, two q-groups of 16 rows each
  FragB qb[2][2];
  #pragma unroll
  for (int g = 0; g < 2; ++g) {
    const __bf16* qrow = Qh + (size_t)(b * Ss + q0 + g * 16 + l15) * Ee + h * Dd;
    #pragma unroll
    for (int f = 0; f < 2; ++f) {
      const __bf16* p = qrow + f * 32 + hi * 8;
      qb[g][f].q[0] = *(const uint4*)(p);
      qb[g][f].q[1] = *(const uint4*)(p + 16);
    }
  }

  const __bf16* Kbase = Kh + (size_t)(b * Ss) * Ee + h * Dd;
  const __bf16* Vbase = Vh + (size_t)(b * Ss) * Ee + h * Dd;

  float m[2] = {-1e30f, -1e30f};
  float l[2] = {0.f, 0.f};

  // ---------------- Pass 1: online max / sum-exp ----------------
  for (int kt = 0; kt < Ss / 32; ++kt) {
    #pragma unroll
    for (int s = 0; s < 2; ++s) {
      const __bf16* kr = Kbase + (size_t)(kt * 32 + s * 16 + l15) * Ee;
      FragF c[2];
      #pragma unroll
      for (int g = 0; g < 2; ++g)
        #pragma unroll
        for (int i = 0; i < 8; ++i) c[g].f[i] = 0.f;
      #pragma unroll
      for (int f = 0; f < 2; ++f) {
        FragB ka;
        const __bf16* p = kr + f * 32 + hi * 8;
        ka.q[0] = *(const uint4*)(p);
        ka.q[1] = *(const uint4*)(p + 16);
        c[0].v = wmma_bf16(ka.v, qb[0][f].v, c[0].v);
        c[1].v = wmma_bf16(ka.v, qb[1][f].v, c[1].v);
      }
      #pragma unroll
      for (int g = 0; g < 2; ++g) {
        float tmax = -1e30f;
        #pragma unroll
        for (int i = 0; i < 8; ++i) {
          c[g].f[i] *= 0.125f;                  // 1/sqrt(64)
          tmax = fmaxf(tmax, c[g].f[i]);
        }
        tmax = fmaxf(tmax, __shfl_xor(tmax, 16, 32));
        const float mn = fmaxf(m[g], tmax);
        float acc = 0.f;
        #pragma unroll
        for (int i = 0; i < 8; ++i) acc += __expf(c[g].f[i] - mn);
        acc += __shfl_xor(acc, 16, 32);
        l[g] = l[g] * __expf(m[g] - mn) + acc;
        m[g] = mn;
      }
    }
  }
  const float linv0 = 1.0f / l[0];
  const float linv1 = 1.0f / l[1];

  FragF ctx[2][4];
  #pragma unroll
  for (int g = 0; g < 2; ++g)
    #pragma unroll
    for (int c = 0; c < 4; ++c)
      #pragma unroll
      for (int i = 0; i < 8; ++i) ctx[g][c].f[i] = 0.f;

  float* arow0 = attn + (size_t)((b * Hh + h) * Ss + q0 + l15) * Ss;
  float* arow1 = arow0 + (size_t)16 * Ss;

  // ---------------- Pass 2: attn write + ctx accumulate ----------------
  for (int kt = 0; kt < Ss / 32; ++kt) {
    __syncthreads();
    {
      // cooperative transposed staging of V tile: 128 threads, each 16 elems
      const int t   = threadIdx.x;
      const int key = t >> 2;          // 0..31
      const int ds  = (t & 3) * 16;    // 0,16,32,48
      const __bf16* vp = Vbase + (size_t)(kt * 32 + key) * Ee + ds;
      union { uint4 q; __bf16 e[8]; } u0, u1;
      u0.q = *(const uint4*)(vp);
      u1.q = *(const uint4*)(vp + 8);
      #pragma unroll
      for (int i = 0; i < 8; ++i) {
        Vt[ds + i][key]     = u0.e[i];
        Vt[ds + 8 + i][key] = u1.e[i];
      }
    }
    __syncthreads();

    FragB ap[2];   // probabilities as A-fragments for ctx WMMA (K = 32 keys)
    #pragma unroll
    for (int s = 0; s < 2; ++s) {
      const __bf16* kr = Kbase + (size_t)(kt * 32 + s * 16 + l15) * Ee;
      FragF c[2];
      #pragma unroll
      for (int g = 0; g < 2; ++g)
        #pragma unroll
        for (int i = 0; i < 8; ++i) c[g].f[i] = 0.f;
      #pragma unroll
      for (int f = 0; f < 2; ++f) {
        FragB ka;
        const __bf16* p = kr + f * 32 + hi * 8;
        ka.q[0] = *(const uint4*)(p);
        ka.q[1] = *(const uint4*)(p + 16);
        c[0].v = wmma_bf16(ka.v, qb[0][f].v, c[0].v);
        c[1].v = wmma_bf16(ka.v, qb[1][f].v, c[1].v);
      }
      #pragma unroll
      for (int g = 0; g < 2; ++g) {
        const float linv = g ? linv1 : linv0;
        #pragma unroll
        for (int i = 0; i < 8; ++i) {
          const float pv = __expf(c[g].f[i] * 0.125f - m[g]) * linv;
          c[g].f[i] = pv;
          ap[g].e[s * 8 + i] = (__bf16)pv;  // C rows land exactly on A K-dim
        }
        // write normalized attention: 8 contiguous floats per lane
        float4* dst = (float4*)((g ? arow1 : arow0) + kt * 32 + s * 16 + hi * 8);
        dst[0] = *(float4*)&c[g].f[0];
        dst[1] = *(float4*)&c[g].f[4];
      }
    }

    #pragma unroll
    for (int cc = 0; cc < 4; ++cc) {
      FragB vb;   // B_v: lane=col d (cc*16+l15), keys {0..7,16..23} / {8..15,24..31} from LDS
      const __bf16* vtp = &Vt[cc * 16 + l15][hi * 8];
      vb.q[0] = *(const uint4*)(vtp);
      vb.q[1] = *(const uint4*)(vtp + 16);
      ctx[0][cc].v = wmma_bf16(ap[0].v, vb.v, ctx[0][cc].v);
      ctx[1][cc].v = wmma_bf16(ap[1].v, vb.v, ctx[1][cc].v);
    }
  }

  // write ctx (bf16, [B,S,E] with head offset) for the output projection
  #pragma unroll
  for (int g = 0; g < 2; ++g)
    #pragma unroll
    for (int cc = 0; cc < 4; ++cc) {
      const int col = h * Dd + cc * 16 + l15;
      #pragma unroll
      for (int r = 0; r < 8; ++r) {
        const int qq = q0 + g * 16 + r + hi * 8;
        ctxout[(size_t)(b * Ss + qq) * Ee + col] = (__bf16)ctx[g][cc].f[r];
      }
    }
}

// ---------------------------------------------------------------------------
extern "C" void kernel_launch(void* const* d_in, const int* in_sizes, int n_in,
                              void* d_out, int out_size, void* d_ws, size_t ws_size,
                              hipStream_t stream) {
  (void)in_sizes; (void)n_in; (void)out_size; (void)ws_size;
  const float* v  = (const float*)d_in[0];
  const float* k  = (const float*)d_in[1];
  const float* q  = (const float*)d_in[2];
  const float* Wq = (const float*)d_in[3];
  const float* bq = (const float*)d_in[4];
  const float* Wk = (const float*)d_in[5];
  const float* bk = (const float*)d_in[6];
  const float* Wv = (const float*)d_in[7];
  const float* bv = (const float*)d_in[8];
  const float* Wo = (const float*)d_in[9];
  const float* bo = (const float*)d_in[10];

  char* ws = (char*)d_ws;
  const size_t MB = 1024u * 1024u;
  __bf16* Wtq = (__bf16*)(ws + 0 * MB);
  __bf16* Wtk = (__bf16*)(ws + 2 * MB);
  __bf16* Wtv = (__bf16*)(ws + 4 * MB);
  __bf16* Wto = (__bf16*)(ws + 6 * MB);
  __bf16* Qh  = (__bf16*)(ws + 8 * MB);
  __bf16* Khb = (__bf16*)(ws + 24 * MB);
  __bf16* Vhb = (__bf16*)(ws + 40 * MB);
  __bf16* Ctx = (__bf16*)(ws + 56 * MB);

  wt_kernel<<<4096, 256, 0, stream>>>(Wq, Wtq);
  wt_kernel<<<4096, 256, 0, stream>>>(Wk, Wtk);
  wt_kernel<<<4096, 256, 0, stream>>>(Wv, Wtv);
  wt_kernel<<<4096, 256, 0, stream>>>(Wo, Wto);

  dim3 gg(Bb * Ss / 64, Ee / 64);   // (128, 16)
  gemm_qkv_kernel<<<gg, 128, 0, stream>>>(q, Wtq, bq, Qh);
  gemm_qkv_kernel<<<gg, 128, 0, stream>>>(k, Wtk, bk, Khb);
  gemm_qkv_kernel<<<gg, 128, 0, stream>>>(v, Wtv, bv, Vhb);

  float* out  = (float*)d_out;
  float* attn = out + (size_t)Bb * Ss * Ee;
  dim3 ga(Ss / 128, Hh, Bb);          // (16, 16, 4)
  attn_kernel<<<ga, 128, 0, stream>>>(Qh, Khb, Vhb, attn, Ctx);

  gemm_out_kernel<<<gg, 128, 0, stream>>>(Ctx, Wto, bo, out);
}
